// conv_block_20830591386251
// MI455X (gfx1250) — matrix operand order repaired and checked by reading, not compiled, not run on hardware
//
#include <hip/hip_runtime.h>
#include <hip/hip_bf16.h>

typedef __attribute__((ext_vector_type(16))) _Float16 v16h;
typedef __attribute__((ext_vector_type(8)))  _Float16 v8h;
typedef __attribute__((ext_vector_type(8)))  float    v8f;
typedef __attribute__((ext_vector_type(4)))  int      v4i;
typedef __attribute__((address_space(1)))    v4i      gv4i;   // HIP prints as "__device__"
typedef __attribute__((address_space(3)))    v4i      sv4i;   // HIP prints as "__shared__"

#define EPS     1e-7f
#define BN_EPS  1e-5f

// Problem geometry (fixed by the reference)
#define BATCH 128
#define CIN   64
#define COUT  64
#define HH    80
#define WW    80
#define G     16
#define CC    4
#define H2    40
#define W2    40
#define NPIX  (BATCH*HH*WW)  // per-channel reduce count = 819200

// Workspace layout (bytes, all 256-aligned)
#define XH_OFF    0ull                                  // f16 NHWC x: 104,857,600
#define WPK_OFF   104857600ull                          // packed f16 weights: 73,728
#define CONV_OFF  104931328ull                          // fp32 conv out NCHW: 209,715,200
#define STATS_OFF 314646528ull                          // sum[64] | sumsq[64]
#define SS_OFF    314647040ull                          // scale[64] | shift[64]
#define LOCS_OFF  314647552ull                          // loc_e[16*4] | loc_l[16*4]

#if __has_builtin(__builtin_amdgcn_global_load_async_to_lds_b128)
#define HAVE_ASYNC_LDS 1
#endif

// ---------------------------------------------------------------------------
// Kernel 1: x fp32 NCHW -> f16 NHWC  (one block per (b,h) row)
// ---------------------------------------------------------------------------
__global__ __launch_bounds__(256) void xpose_kernel(const float* __restrict__ x,
                                                    _Float16* __restrict__ xh) {
  __shared__ float tile[WW * 65];  // [w][c], pad 65 to avoid bank conflicts
  const int wg = blockIdx.x;
  const int b = wg / HH, h = wg % HH;
  const int tid = threadIdx.x;
  for (int i = tid; i < CIN * WW; i += 256) {
    int c = i / WW, w = i % WW;  // consecutive tid -> consecutive w (coalesced)
    tile[w * 65 + c] =
        __builtin_nontemporal_load(&x[(((size_t)b * CIN + c) * HH + h) * WW + w]);
  }
  __syncthreads();
  _Float16* dst = &xh[((size_t)(b * HH + h)) * WW * CIN];
  for (int i = tid; i < CIN * WW; i += 256) {
    int w = i / CIN, c = i % CIN;  // consecutive tid -> consecutive c (coalesced)
    dst[i] = (_Float16)tile[w * 65 + c];
  }
}

// ---------------------------------------------------------------------------
// Kernel 2: pack weights into per-lane WMMA A-fragment layout; zero BN stats.
// A layout (16-bit 16x32): lane<16 -> M=lane, K-halves {kb..kb+7, 16+kb..16+kb+7}
// with kb = (lane>>4)*8.  Packed so each lane's 16 halves are contiguous.
// ---------------------------------------------------------------------------
__global__ __launch_bounds__(256) void pack_w_kernel(const float* __restrict__ cw,
                                                     _Float16* __restrict__ wpk,
                                                     float* __restrict__ stats) {
  const int tid = threadIdx.x;
  if (tid < 128) stats[tid] = 0.0f;  // sum[64] | sumsq[64] zeroed every launch
  // 18 kchunks * 4 mtiles * 32 lanes * 16 halves = 36864 entries
  for (int i = tid; i < 36864; i += 256) {
    int j = i & 15;
    int lane = (i >> 4) & 31;
    int mt = (i >> 9) & 3;
    int kchunk = i >> 11;  // 0..17, kchunk = tap*2 + kc
    int l15 = lane & 15, kb = (lane >> 4) * 8;
    int m = mt * 16 + l15;
    int klocal = (j < 8) ? (kb + j) : (16 + kb + (j - 8));
    int tap = kchunk >> 1;
    int cin = ((kchunk & 1) << 5) + klocal;
    wpk[i] = (_Float16)cw[(m * CIN + cin) * 9 + tap];
  }
}

// ---------------------------------------------------------------------------
// Kernel 3: implicit-GEMM conv via WMMA.
// Workgroup = one (b,h) output row. 4 waves = 4 cout tiles of 16.
// Each wave: 5 N-subtiles of 16 pixels covering w=0..79, acc in v8f each.
// K loop: 9 taps * 2 chunks of 32 input channels.
// ---------------------------------------------------------------------------
#define XSTR 72  // padded channel stride in LDS x tile (halves)
__global__ __launch_bounds__(128) void conv_wmma_kernel(
    const _Float16* __restrict__ xh,   // NHWC f16
    const _Float16* __restrict__ wpk,  // packed A fragments
    const float* __restrict__ bias,
    float* __restrict__ convout,       // NCHW fp32
    float* __restrict__ gsum, float* __restrict__ gsqr) {
  __shared__ _Float16 xt[3 * 82 * XSTR];  // [row 0..2][col 0..81][c 0..63 (+pad)]
  __shared__ float sSum[COUT], sSqr[COUT];

  const int wg = blockIdx.x;
  const int b = wg / HH, h = wg % HH;
  const int tid = threadIdx.x;
  const int lane = tid & 31;
  const int wave = tid >> 5;  // M tile
  const int l15 = lane & 15;
  const int lhalf = lane >> 4;

  if (tid < COUT) { sSum[tid] = 0.0f; sSqr[tid] = 0.0f; }

  // Stage input tile: 3 rows x 82 cols x 64 ch, zero-padded halo.
  for (int i = tid; i < 3 * 82 * 8; i += 128) {
    int chunk = i & 7;  // 8 x 16B per column
    int col = (i >> 3) % 82;
    int r = (i >> 3) / 82;
    int hy = h + r - 1, w = col - 1;
    const bool inb = (hy >= 0 && hy < HH && w >= 0 && w < WW);
    const size_t didx = (size_t)(r * 82 + col) * XSTR + chunk * 8;
    if (inb) {
      const size_t sidx = (((size_t)(b * HH + hy)) * WW + w) * CIN + chunk * 8;
#ifdef HAVE_ASYNC_LDS
      __builtin_amdgcn_global_load_async_to_lds_b128(
          (gv4i*)&xh[sidx], (sv4i*)&xt[didx], 0, 0);
#else
      *(v8h*)&xt[didx] = *(const v8h*)&xh[sidx];
#endif
    } else {
      v8h z;
#pragma unroll
      for (int q = 0; q < 8; ++q) z[q] = (_Float16)0.0f;
      *(v8h*)&xt[didx] = z;
    }
  }
#ifdef HAVE_ASYNC_LDS
#if __has_builtin(__builtin_amdgcn_s_wait_asynccnt)
  __builtin_amdgcn_s_wait_asynccnt(0);
#endif
#endif
  __syncthreads();

  v8f acc[5];
#pragma unroll
  for (int n = 0; n < 5; ++n)
#pragma unroll
    for (int q = 0; q < 8; ++q) acc[n][q] = 0.0f;

  union BFrag { v16h v; v8h half8[2]; };

#pragma unroll 1
  for (int tap = 0; tap < 9; ++tap) {
    const int dy = tap / 3, dx = tap % 3;
#pragma unroll
    for (int kc = 0; kc < 2; ++kc) {
      const int kchunk = tap * 2 + kc;
      // A fragment: 16 contiguous halves per lane, straight from L2
      const v16h a = *(const v16h*)&wpk[((size_t)(kchunk * 4 + wave) * 32 + lane) * 16];
      if (kchunk + 1 < 18)  // prefetch next A fragment -> global_prefetch_b8
        __builtin_prefetch(&wpk[((size_t)((kchunk + 1) * 4 + wave) * 32 + lane) * 16], 0, 1);
#pragma unroll
      for (int nt = 0; nt < 5; ++nt) {
        const int col = nt * 16 + l15 + dx;  // 0..81
        // B fragment: lane holds N=l15, K = kc*32 + {lhalf*8 + j, 16 + lhalf*8 + j}
        const _Float16* bp = &xt[(dy * 82 + col) * XSTR + kc * 32 + lhalf * 8];
        BFrag bf;
        bf.half8[0] = *(const v8h*)bp;
        bf.half8[1] = *(const v8h*)(bp + 16);
        acc[nt] = __builtin_amdgcn_wmma_f32_16x16x32_f16(
            false, a, false, bf.v, (short)0, acc[nt], false, false);
      }
    }
  }

  // Epilogue: bias add, NT store fp32, per-channel partial sums.
#pragma unroll
  for (int r = 0; r < 8; ++r) {
    const int cout = wave * 16 + r + lhalf * 8;
    const float bv = bias[cout];
    float p1 = 0.0f, p2 = 0.0f;
#pragma unroll
    for (int nt = 0; nt < 5; ++nt) {
      const float val = acc[nt][r] + bv;
      const int w = nt * 16 + l15;
      __builtin_nontemporal_store(
          val, &convout[(((size_t)b * COUT + cout) * HH + h) * WW + w]);
      p1 += val;
      p2 += val * val;
    }
    // butterfly reduce across the 16 lanes sharing this cout (wave32 halves)
#pragma unroll
    for (int off = 1; off < 16; off <<= 1) {
      p1 += __shfl_xor(p1, off, 32);
      p2 += __shfl_xor(p2, off, 32);
    }
    if (l15 == 0) {
      atomicAdd(&sSum[cout], p1);  // ds_add_f32, 2 lanes/wave only
      atomicAdd(&sSqr[cout], p2);
    }
  }
  __syncthreads();
  if (tid < COUT) {
    atomicAdd(&gsum[tid], sSum[tid]);
    atomicAdd(&gsqr[tid], sSqr[tid]);
  }
}

// ---------------------------------------------------------------------------
// Kernel 4: finalize BN scale/shift and hyperbolic loc vectors (expmap0).
// ---------------------------------------------------------------------------
__global__ __launch_bounds__(64) void finalize_kernel(
    const float* __restrict__ stats, const float* __restrict__ gamma,
    const float* __restrict__ beta, const float* __restrict__ curvature,
    const float* __restrict__ loc_exp, const float* __restrict__ loc_log,
    float* __restrict__ ss, float* __restrict__ locs) {
  const int t = threadIdx.x;
  if (t < COUT) {
    const float invN = 1.0f / (float)NPIX;
    const float mean = stats[t] * invN;
    const float var = stats[COUT + t] * invN - mean * mean;
    const float sc = gamma[t] * rsqrtf(var + BN_EPS);
    ss[t] = sc;
    ss[COUT + t] = beta[t] - mean * sc;
  }
  if (t < G) {
    const float k = curvature[t];
    const float sk = sqrtf(k);
#pragma unroll
    for (int which = 0; which < 2; ++which) {
      const float* src = (which == 0) ? &loc_exp[t * CC] : &loc_log[t * CC];
      float u[CC], n2 = 0.0f;
#pragma unroll
      for (int j = 0; j < CC; ++j) { u[j] = src[j]; n2 += u[j] * u[j]; }
      const float n = fmaxf(sqrtf(n2), EPS);
      const float f = tanhf(sk * n) / (sk * n);
#pragma unroll
      for (int j = 0; j < CC; ++j) locs[which * 64 + t * CC + j] = f * u[j];
    }
  }
}

// ---------------------------------------------------------------------------
// Kernel 5: fused BN + ReLU + 2x2 maxpool + expmap/logmap.
// One thread per (b, g, h2, w2) output 4-vector.
// ---------------------------------------------------------------------------
__device__ __forceinline__ void mobius_add4(const float x[CC], const float y[CC],
                                            float k, float out[CC]) {
  float x2 = 0.0f, y2 = 0.0f, xy = 0.0f;
#pragma unroll
  for (int j = 0; j < CC; ++j) { x2 += x[j] * x[j]; y2 += y[j] * y[j]; xy += x[j] * y[j]; }
  const float ca = 1.0f + 2.0f * k * xy + k * y2;
  const float cb = 1.0f - k * x2;
  const float den = fmaxf(1.0f + 2.0f * k * xy + k * k * x2 * y2, EPS);
  const float rden = 1.0f / den;
#pragma unroll
  for (int j = 0; j < CC; ++j) out[j] = (ca * x[j] + cb * y[j]) * rden;
}

__global__ __launch_bounds__(256) void post_kernel(
    const float* __restrict__ conv, const float* __restrict__ ss,
    const float* __restrict__ locs, const float* __restrict__ curvature,
    float* __restrict__ out) {
  const int idx = blockIdx.x * 256 + threadIdx.x;  // B*G*H2*W2 = 3,276,800 total
  const int w2 = idx % W2;
  int t = idx / W2;
  const int h2 = t % H2; t /= H2;
  const int g = t % G;
  const int b = t / G;

  const float k = curvature[g];
  const float sk = sqrtf(k);
  float xe[CC], xl[CC], v[CC];
#pragma unroll
  for (int j = 0; j < CC; ++j) {
    xe[j] = locs[g * CC + j];
    xl[j] = locs[64 + g * CC + j];
  }

  // BN + ReLU + 2x2 maxpool for the 4 channels of this group
#pragma unroll
  for (int j = 0; j < CC; ++j) {
    const int c = g * CC + j;
    const float* p = &conv[(((size_t)b * COUT + c) * HH + 2 * h2) * WW + 2 * w2];
    const float sc = ss[c], sh = ss[COUT + c];
    const float a0 = __builtin_nontemporal_load(p + 0);
    const float a1 = __builtin_nontemporal_load(p + 1);
    const float a2 = __builtin_nontemporal_load(p + WW);
    const float a3 = __builtin_nontemporal_load(p + WW + 1);
    float m = fmaxf(fmaxf(a0 * sc + sh, a1 * sc + sh),
                    fmaxf(a2 * sc + sh, a3 * sc + sh));
    v[j] = fmaxf(m, 0.0f);
  }

  // expmap(loc_e, v, k)
  float vn2 = 0.0f, xe2 = 0.0f;
#pragma unroll
  for (int j = 0; j < CC; ++j) { vn2 += v[j] * v[j]; xe2 += xe[j] * xe[j]; }
  const float un = fmaxf(sqrtf(vn2), EPS);
  const float lam = 2.0f / fmaxf(1.0f - k * xe2, EPS);
  const float tfac = tanhf(sk * lam * un * 0.5f) / (sk * un);
  float sec[CC];
#pragma unroll
  for (int j = 0; j < CC; ++j) sec[j] = tfac * v[j];
  float y[CC];
  mobius_add4(xe, sec, k, y);

  // logmap(loc_l, y, k)
  float nxl[CC];
#pragma unroll
  for (int j = 0; j < CC; ++j) nxl[j] = -xl[j];
  float sub[CC];
  mobius_add4(nxl, y, k, sub);
  float sn2 = 0.0f, xl2 = 0.0f;
#pragma unroll
  for (int j = 0; j < CC; ++j) { sn2 += sub[j] * sub[j]; xl2 += xl[j] * xl[j]; }
  const float sn = fmaxf(sqrtf(sn2), EPS);
  const float lam2 = 2.0f / fmaxf(1.0f - k * xl2, EPS);
  float z = sk * sn;
  z = fminf(fmaxf(z, -1.0f + EPS), 1.0f - EPS);
  const float at = 0.5f * (log1pf(z) - log1pf(-z));
  const float fac = (2.0f / (sk * lam2)) * at / sn;

#pragma unroll
  for (int j = 0; j < CC; ++j)
    __builtin_nontemporal_store(
        fac * sub[j], &out[(((size_t)b * COUT + g * CC + j) * H2 + h2) * W2 + w2]);
}

// ---------------------------------------------------------------------------
extern "C" void kernel_launch(void* const* d_in, const int* in_sizes, int n_in,
                              void* d_out, int out_size, void* d_ws, size_t ws_size,
                              hipStream_t stream) {
  const float* x         = (const float*)d_in[0];
  const float* curvature = (const float*)d_in[1];
  const float* conv_w    = (const float*)d_in[2];
  const float* conv_b    = (const float*)d_in[3];
  const float* gamma     = (const float*)d_in[4];
  const float* beta      = (const float*)d_in[5];
  const float* loc_exp   = (const float*)d_in[6];
  const float* loc_log   = (const float*)d_in[7];
  float* out = (float*)d_out;

  char* ws = (char*)d_ws;
  _Float16* xh  = (_Float16*)(ws + XH_OFF);
  _Float16* wpk = (_Float16*)(ws + WPK_OFF);
  float* convbuf = (float*)(ws + CONV_OFF);
  float* stats   = (float*)(ws + STATS_OFF);
  float* ss      = (float*)(ws + SS_OFF);
  float* locs    = (float*)(ws + LOCS_OFF);

  xpose_kernel<<<BATCH * HH, 256, 0, stream>>>(x, xh);
  pack_w_kernel<<<1, 256, 0, stream>>>(conv_w, wpk, stats);
  conv_wmma_kernel<<<BATCH * HH, 128, 0, stream>>>(xh, wpk, conv_b, convbuf,
                                                   stats, stats + COUT);
  finalize_kernel<<<1, 64, 0, stream>>>(stats, gamma, beta, curvature,
                                        loc_exp, loc_log, ss, locs);
  post_kernel<<<(BATCH * G * H2 * W2) / 256, 256, 0, stream>>>(convbuf, ss, locs,
                                                               curvature, out);
}
